// SwapGNN_23845658428317
// MI455X (gfx1250) — matrix operand order repaired and checked by reading, not compiled, not run on hardware
//
#include <hip/hip_runtime.h>
#include <hip/hip_bf16.h>

#define HEADS 4
#define CH 32
#define DIM 128
#define FCDIM 256
#define NEG_SLOPE 0.2f

typedef __attribute__((ext_vector_type(16))) __bf16 bf16x16;
typedef __attribute__((ext_vector_type(8)))  float  f32x8;

union BfFrag { bf16x16 v; uint4 q[2]; };

#if __has_builtin(__builtin_amdgcn_cvt_pk_bf16_f32)
typedef __attribute__((ext_vector_type(2))) __bf16 bf16x2;
__device__ __forceinline__ unsigned pack2bf(float lo, float hi) {
  bf16x2 v = __builtin_amdgcn_cvt_pk_bf16_f32(lo, hi);
  unsigned r; __builtin_memcpy(&r, &v, 4); return r;
}
#else
__device__ __forceinline__ unsigned f2bf_u(float f) {
  union { float f; unsigned u; } x; x.f = f;
  unsigned r = x.u + 0x7FFFu + ((x.u >> 16) & 1u);   // round-to-nearest-even
  return r >> 16;
}
__device__ __forceinline__ unsigned pack2bf(float lo, float hi) {
  return f2bf_u(lo) | (f2bf_u(hi) << 16);
}
#endif

// out[M,Nout] = A[M,K] @ W[Nout,K]^T (+bias, +relu) via v_wmma_f32_16x16x32_bf16.
// 128 threads = 4 waves. Workgroup tile 64x64; each wave computes a 32x32 quadrant
// (2 A-frags x 2 B-frags = 4 WMMAs per 32-deep K step). Double-buffered LDS in
// fragment-major order: stage step s+1 while step s is consumed; each lane fetches
// a fragment with two ds_load_b128. Tile loads are branch-free clamped b64 loads.
__global__ void __launch_bounds__(128)
gemm_bf16wmma(const float* __restrict__ A, const float* __restrict__ W,
              const float* __restrict__ bias, float* __restrict__ out,
              int M, int K, int Nout, int relu) {
  // [buf 0..1][rowgroup 0..3][lane 0..31][slot 0..7] packed bf16 pairs
  __shared__ unsigned AsF[2][4 * 32 * 8];
  __shared__ unsigned BsF[2][4 * 32 * 8];

  const int tileM = blockIdx.x * 64;
  const int tileN = blockIdx.y * 64;
  const int t    = threadIdx.x;      // 0..127
  const int lane = t & 31;
  const int wave = t >> 5;           // 0..3
  const int rb   = (wave >> 1) * 32; // wave's 32-row block within tile
  const int cb   = (wave & 1) * 32;  // wave's 32-col block within tile
  const int lm   = lane & 15;
  const int lh   = lane >> 4;

  // Precompute this thread's 8 staging slots (independent of k0).
  int aRow[8], aK[8], aSlot[8];      // A: 64 rows x 16 k-pairs
  int bCol[8], bK[8], bSlot[8];      // B: 16 k-pairs x 64 cols
  #pragma unroll
  for (int pp = 0; pp < 8; ++pp) {
    int p = t + 128 * pp;            // 0..1023
    int r = p >> 4;                  // row 0..63
    int k = (p & 15) << 1;           // even k 0..30
    aRow[pp] = r; aK[pp] = k;
    int alan = ((k >> 3) & 1) * 16 + (r & 15);
    int ae   = ((k >> 4) << 3) + (k & 7);
    aSlot[pp] = ((r >> 4) * 32 + alan) * 8 + (ae >> 1);
    int c  = p & 63;                 // col 0..63
    int kb = (p >> 6) << 1;          // even k 0..30
    bCol[pp] = c; bK[pp] = kb;
    int blan = ((kb >> 3) & 1) * 16 + (c & 15);
    bSlot[pp] = ((c >> 4) * 32 + blan) * 8 + ((((kb >> 4) << 3) + (kb & 7)) >> 1);
  }

  auto stage = [&](int k0, int buf) {
    float2 tA[8], tB[8];
    // issue all 16 global_load_b64 first (clamped rows -> no exec divergence)
    #pragma unroll
    for (int pp = 0; pp < 8; ++pp) {
      int gm = tileM + aRow[pp];
      int gmc = (gm < M) ? gm : (M - 1);
      tA[pp] = *(const float2*)&A[(size_t)gmc * K + (k0 + aK[pp])];
    }
    #pragma unroll
    for (int pp = 0; pp < 8; ++pp) {
      tB[pp] = *(const float2*)&W[(size_t)(tileN + bCol[pp]) * K + (k0 + bK[pp])];
    }
    // convert + scatter to fragment-major LDS
    #pragma unroll
    for (int pp = 0; pp < 8; ++pp) {
      unsigned pk = pack2bf(tA[pp].x, tA[pp].y);
      AsF[buf][aSlot[pp]] = ((tileM + aRow[pp]) < M) ? pk : 0u;
    }
    #pragma unroll
    for (int pp = 0; pp < 8; ++pp) {
      BsF[buf][bSlot[pp]] = pack2bf(tB[pp].x, tB[pp].y);
    }
  };

  f32x8 acc00 = {}, acc01 = {}, acc10 = {}, acc11 = {};

  const int nsteps = K >> 5;
  stage(0, 0);
  __syncthreads();

  for (int s = 0; s < nsteps; ++s) {
    const int buf = s & 1;
    // fragment fetch: contiguous 32B per lane -> 2x ds_load_b128 each
    BfFrag a0, a1, b0, b1;
    const uint4* pa0 = (const uint4*)&AsF[buf][(((rb >> 4) + 0) * 32 + lane) * 8];
    const uint4* pa1 = (const uint4*)&AsF[buf][(((rb >> 4) + 1) * 32 + lane) * 8];
    const uint4* pb0 = (const uint4*)&BsF[buf][(((cb >> 4) + 0) * 32 + lane) * 8];
    const uint4* pb1 = (const uint4*)&BsF[buf][(((cb >> 4) + 1) * 32 + lane) * 8];
    a0.q[0] = pa0[0]; a0.q[1] = pa0[1];
    a1.q[0] = pa1[0]; a1.q[1] = pa1[1];
    b0.q[0] = pb0[0]; b0.q[1] = pb0[1];
    b1.q[0] = pb1[0]; b1.q[1] = pb1[1];

    if (s + 1 < nsteps) stage((s + 1) * 32, buf ^ 1);  // overlap next-step staging

    acc00 = __builtin_amdgcn_wmma_f32_16x16x32_bf16(false, a0.v, false, b0.v,
                                                    (short)0, acc00, false, false);
    acc01 = __builtin_amdgcn_wmma_f32_16x16x32_bf16(false, a0.v, false, b1.v,
                                                    (short)0, acc01, false, false);
    acc10 = __builtin_amdgcn_wmma_f32_16x16x32_bf16(false, a1.v, false, b0.v,
                                                    (short)0, acc10, false, false);
    acc11 = __builtin_amdgcn_wmma_f32_16x16x32_bf16(false, a1.v, false, b1.v,
                                                    (short)0, acc11, false, false);
    __syncthreads();
  }

  // ---- epilogue: C/D layout m = 16*sub + lh*8 + r, n = lm ----
  #pragma unroll
  for (int i = 0; i < 2; ++i) {
    #pragma unroll
    for (int j = 0; j < 2; ++j) {
      const f32x8& acc = (i == 0) ? (j == 0 ? acc00 : acc01)
                                  : (j == 0 ? acc10 : acc11);
      int n = tileN + cb + j * 16 + lm;
      float bv = bias ? bias[n] : 0.f;
      #pragma unroll
      for (int r = 0; r < 8; ++r) {
        int m = tileM + rb + i * 16 + (lh << 3) + r;
        if (m < M) {
          float v = acc[r] + bv;
          if (relu) v = fmaxf(v, 0.0f);
          out[(size_t)m * Nout + n] = v;
        }
      }
    }
  }
}

__global__ void att_coeff_kernel(const float* __restrict__ feat,
                                 const float* __restrict__ a_src,
                                 const float* __restrict__ a_dst,
                                 float* __restrict__ aS, float* __restrict__ aD, int Nn) {
  int id = blockIdx.x * blockDim.x + threadIdx.x;
  if (id >= Nn * HEADS) return;
  int n = id >> 2, h = id & 3;
  const float* f  = feat + (size_t)n * DIM + h * CH;
  const float* as = a_src + h * CH;
  const float* ad = a_dst + h * CH;
  float s = 0.f, d = 0.f;
  #pragma unroll
  for (int c = 0; c < CH; ++c) { float fv = f[c]; s += fv * as[c]; d += fv * ad[c]; }
  aS[id] = s; aD[id] = d;
}

__global__ void seg_init_kernel(float* segmax, float* segsum, int count) {
  int i = blockIdx.x * blockDim.x + threadIdx.x;
  if (i < count) { segmax[i] = -__builtin_inff(); segsum[i] = 0.f; }
}

__device__ __forceinline__ void atomicMaxF(float* addr, float v) {
  if (v >= 0.f) atomicMax((int*)addr, __float_as_int(v));
  else          atomicMin((unsigned int*)addr, (unsigned int)__float_as_int(v));
}

__global__ void edge_max_kernel(const int* __restrict__ src, const int* __restrict__ dst,
                                const float* __restrict__ aS, const float* __restrict__ aD,
                                float* __restrict__ eBuf, float* __restrict__ segmax,
                                int E, int Nn) {
  int t = blockIdx.x * blockDim.x + threadIdx.x;
  int total = (E + Nn) * HEADS;
  if (t >= total) return;
  int e = t >> 2, h = t & 3;
  int s, d;
  if (e < E) { s = src[e]; d = dst[e]; } else { s = d = e - E; }
  float v = aS[s * HEADS + h] + aD[d * HEADS + h];
  v = (v >= 0.f) ? v : v * NEG_SLOPE;
  eBuf[t] = v;
  atomicMaxF(&segmax[d * HEADS + h], v);
}

__global__ void edge_exp_kernel(const int* __restrict__ dst, float* __restrict__ eBuf,
                                const float* __restrict__ segmax, float* __restrict__ segsum,
                                int E, int Nn) {
  int t = blockIdx.x * blockDim.x + threadIdx.x;
  int total = (E + Nn) * HEADS;
  if (t >= total) return;
  int e = t >> 2, h = t & 3;
  int d = (e < E) ? dst[e] : (e - E);
  float v = __expf(eBuf[t] - segmax[d * HEADS + h]);
  eBuf[t] = v;
  atomicAdd(&segsum[d * HEADS + h], v);
}

__global__ void bias_init_kernel(const float* __restrict__ b, float* __restrict__ out, int Nn) {
  int i = blockIdx.x * blockDim.x + threadIdx.x;
  if (i < Nn * DIM) out[i] = b[i & (DIM - 1)];
}

__global__ void agg_kernel(const int* __restrict__ src, const int* __restrict__ dst,
                           const float* __restrict__ eBuf, const float* __restrict__ segsum,
                           const float* __restrict__ feat, float* __restrict__ out,
                           int E, int Nn) {
  int e = blockIdx.x;                 // one edge per block
  int d = threadIdx.x;                // 0..127 channel
  int s, dd;
  if (e < E) { s = src[e]; dd = dst[e]; } else { s = dd = e - E; }
  if (d == 0) __builtin_prefetch(feat + (size_t)s * DIM, 0, 0);  // global_prefetch_b8
  int h = d >> 5;
  float alpha = eBuf[e * HEADS + h] / segsum[dd * HEADS + h];
  atomicAdd(&out[(size_t)dd * DIM + d], alpha * feat[(size_t)s * DIM + d]);
}

__global__ void relu_kernel(float* x, int count) {
  int i = blockIdx.x * blockDim.x + threadIdx.x;
  if (i < count) x[i] = fmaxf(x[i], 0.f);
}

__global__ void final_lin_kernel(const float* __restrict__ z, const float* __restrict__ wf,
                                 const float* __restrict__ bf, float* __restrict__ removed,
                                 float* __restrict__ out_logits0, int Nn) {
  int n = blockIdx.x * blockDim.x + threadIdx.x;
  if (n >= Nn) return;
  const float* zr = z + (size_t)n * FCDIM;
  float s = bf[0];
  for (int k = 0; k < FCDIM; ++k) s += zr[k] * wf[k];
  removed[n] = s;
  out_logits0[n] = s;
}

__global__ void argmax_kernel(const float* __restrict__ v, int count, int* __restrict__ outIdx) {
  __shared__ float sv[256];
  __shared__ int   si[256];
  int t = threadIdx.x;
  float best = -__builtin_inff(); int bi = 0;
  for (int i = t; i < count; i += 256) {
    float x = v[i];
    if (x > best) { best = x; bi = i; }
  }
  sv[t] = best; si[t] = bi;
  __syncthreads();
  for (int s = 128; s > 0; s >>= 1) {
    if (t < s) {
      if (sv[t + s] > sv[t] || (sv[t + s] == sv[t] && si[t + s] < si[t])) {
        sv[t] = sv[t + s]; si[t] = si[t + s];
      }
    }
    __syncthreads();
  }
  if (t == 0) *outIdx = si[0];
}

__global__ void qproj_kernel(const float* __restrict__ h, const int* __restrict__ a1,
                             const float* __restrict__ pw, const float* __restrict__ pb,
                             float* __restrict__ q) {
  int d = threadIdx.x;                // 128 threads
  const float* hr = h + (size_t)(*a1) * DIM;
  const float* wr = pw + (size_t)d * DIM;
  float s = pb[d];
  for (int k = 0; k < DIM; ++k) s += hr[k] * wr[k];
  q[d] = tanhf(s);
}

__global__ void newscore_kernel(const float* __restrict__ h, const float* __restrict__ q,
                                float* __restrict__ newv, float* __restrict__ out_logits1, int Nn) {
  __shared__ float qs[DIM];
  if (threadIdx.x < DIM) qs[threadIdx.x] = q[threadIdx.x];
  __syncthreads();
  int n = blockIdx.x * blockDim.x + threadIdx.x;
  if (n >= Nn) return;
  const float* hr = h + (size_t)n * DIM;
  float s = 0.f;
  for (int k = 0; k < DIM; ++k) s += hr[k] * qs[k];
  newv[n] = s;
  out_logits1[n] = s;
}

__global__ void write_actions_kernel(const int* a1, const int* a2, float* out) {
  if (threadIdx.x == 0) { out[0] = (float)(*a1); out[1] = (float)(*a2); }
}

extern "C" void kernel_launch(void* const* d_in, const int* in_sizes, int n_in,
                              void* d_out, int out_size, void* d_ws, size_t ws_size,
                              hipStream_t stream) {
  (void)n_in; (void)out_size; (void)ws_size;
  const float* x     = (const float*)d_in[0];
  const int*   ei    = (const int*)d_in[1];
  const float* gatW  = (const float*)d_in[2];
  const float* gatAs = (const float*)d_in[3];
  const float* gatAd = (const float*)d_in[4];
  const float* gatB  = (const float*)d_in[5];
  const float* w0    = (const float*)d_in[6];
  const float* b0    = (const float*)d_in[7];
  const float* wh    = (const float*)d_in[8];
  const float* bh    = (const float*)d_in[9];
  const float* wf    = (const float*)d_in[10];
  const float* bf    = (const float*)d_in[11];
  const float* pw    = (const float*)d_in[12];
  const float* pb    = (const float*)d_in[13];
  float* out = (float*)d_out;

  const int Nn = in_sizes[0] / DIM;
  const int E  = in_sizes[1] / 2;
  const int E2 = E + Nn;                 // + self loops
  const int* srcA = ei;
  const int* dstA = ei + E;

  // carve workspace
  char* p = (char*)d_ws;
  auto alloc = [&](size_t bytes) -> char* {
    char* r = p; p += (bytes + 255) & ~(size_t)255; return r;
  };
  float* feat     = (float*)alloc((size_t)Nn * DIM * 4);
  float* hbuf     = (float*)alloc((size_t)Nn * DIM * 4);
  float* aS       = (float*)alloc((size_t)Nn * HEADS * 4);
  float* aD       = (float*)alloc((size_t)Nn * HEADS * 4);
  float* segmax   = (float*)alloc((size_t)Nn * HEADS * 4);
  float* segsum   = (float*)alloc((size_t)Nn * HEADS * 4);
  float* eBuf     = (float*)alloc((size_t)E2 * HEADS * 4);
  float* zA       = (float*)alloc((size_t)Nn * FCDIM * 4);
  float* zB       = (float*)alloc((size_t)Nn * FCDIM * 4);
  float* removedB = (float*)alloc((size_t)Nn * 4);
  float* newB     = (float*)alloc((size_t)Nn * 4);
  float* qv       = (float*)alloc(DIM * 4);
  int*   a1       = (int*)alloc(16);
  int*   a2       = (int*)alloc(16);

  const int nhB = (Nn * HEADS + 255) / 256;
  const int ndB = (Nn * DIM + 255) / 256;
  const int egB = (E2 * HEADS + 255) / 256;
  dim3 gGat((Nn + 63) / 64, DIM / 64);
  dim3 gFc((Nn + 63) / 64, FCDIM / 64);

  // ---- 4 GAT layers ----
  const float* cur = x;
  for (int l = 0; l < 4; ++l) {
    gemm_bf16wmma<<<gGat, 128, 0, stream>>>(cur, gatW + (size_t)l * DIM * DIM,
                                            nullptr, feat, Nn, DIM, DIM, 0);
    att_coeff_kernel<<<nhB, 256, 0, stream>>>(feat, gatAs + l * HEADS * CH,
                                              gatAd + l * HEADS * CH, aS, aD, Nn);
    seg_init_kernel<<<nhB, 256, 0, stream>>>(segmax, segsum, Nn * HEADS);
    edge_max_kernel<<<egB, 256, 0, stream>>>(srcA, dstA, aS, aD, eBuf, segmax, E, Nn);
    edge_exp_kernel<<<egB, 256, 0, stream>>>(dstA, eBuf, segmax, segsum, E, Nn);
    bias_init_kernel<<<ndB, 256, 0, stream>>>(gatB + l * DIM, hbuf, Nn);
    agg_kernel<<<E2, DIM, 0, stream>>>(srcA, dstA, eBuf, segsum, feat, hbuf, E, Nn);
    if (l < 3) relu_kernel<<<ndB, 256, 0, stream>>>(hbuf, Nn * DIM);
    cur = hbuf;
  }
  // hbuf == h_l1

  // ---- MLP head ----
  gemm_bf16wmma<<<gFc, 128, 0, stream>>>(hbuf, w0, b0, zA, Nn, DIM, FCDIM, 1);
  float* zin = zA; float* zout = zB;
  for (int i = 0; i < 3; ++i) {
    gemm_bf16wmma<<<gFc, 128, 0, stream>>>(zin, wh + (size_t)i * FCDIM * FCDIM,
                                           bh + i * FCDIM, zout, Nn, FCDIM, FCDIM, 1);
    float* tmp = zin; zin = zout; zout = tmp;
  }
  final_lin_kernel<<<(Nn + 255) / 256, 256, 0, stream>>>(zin, wf, bf, removedB, out, Nn);

  // ---- actions (deterministic argmax in place of host-RNG categorical) ----
  argmax_kernel<<<1, 256, 0, stream>>>(removedB, Nn, a1);
  qproj_kernel<<<1, 128, 0, stream>>>(hbuf, a1, pw, pb, qv);
  newscore_kernel<<<(Nn + 255) / 256, 256, 0, stream>>>(hbuf, qv, newB, out + Nn, Nn);
  argmax_kernel<<<1, 256, 0, stream>>>(newB, Nn, a2);
  write_actions_kernel<<<1, 32, 0, stream>>>(a1, a2, out + 2 * (size_t)Nn);
}
